// MemoryEfficientAttention_43250320671349
// MI455X (gfx1250) — matrix operand order repaired and checked by reading, not compiled, not run on hardware
//
#include <hip/hip_runtime.h>
#include <hip/hip_bf16.h>

typedef __attribute__((ext_vector_type(16))) _Float16 v16h;
typedef __attribute__((ext_vector_type(8)))  _Float16 v8h;
typedef __attribute__((ext_vector_type(8)))  float    v8f;

// ---------------- problem constants ----------------
#define BATCH 2
#define NTOK  4096
#define CDIM  512
#define NHEAD 4
#define HDIM  128
#define MROWS (BATCH * NTOK)      // 8192
#define QKVN  (3 * CDIM)          // 1536

// exp2 on the native v_exp_f32 path
#define EXP2F(x) __builtin_amdgcn_exp2f(x)

// ---------------- fragment loaders (CDNA5 WMMA layouts) ----------------
// A-matrix 16x32 f16 fragment from row-major [*, ld] at tile origin `base`
// (base = A + m0*ld + k0). Per ISA 05_wmma: lanes0-15: elems0-7=K0..7,
// elems8-15=K16..23 ; lanes16-31: elems0-7=K8..15, elems8-15=K24..31.
__device__ __forceinline__ v16h load_a_frag(const _Float16* base, int ld, int lane) {
    const int half = lane >> 4, lr = lane & 15;
    const _Float16* rp = base + (size_t)lr * ld + half * 8;
    v8h lo = *(const v8h*)(rp);
    v8h hi = *(const v8h*)(rp + 16);
    v16h f;
#pragma unroll
    for (int i = 0; i < 8; ++i) { f[i] = lo[i]; f[i + 8] = hi[i]; }
    return f;
}

// B-matrix 32x16 f16 fragment where B[k][n] = Bt[n][k] and Bt is row-major
// [*, ld] (base = Bt + n0*ld + k0). Lanes0-15: elems0-7=K0..7, elems8-15=K16..23;
// lanes16-31: elems0-7=K8..15, elems8-15=K24..31, column n = n0 + lane%16.
__device__ __forceinline__ v16h load_b_frag(const _Float16* base, int ld, int lane) {
    const int half = lane >> 4, lr = lane & 15;
    const _Float16* rp = base + (size_t)lr * ld + half * 8;
    v8h lo = *(const v8h*)(rp);
    v8h hi = *(const v8h*)(rp + 16);
    v16h f;
#pragma unroll
    for (int i = 0; i < 8; ++i) { f[i] = lo[i]; f[i + 8] = hi[i]; }
    return f;
}

__device__ __forceinline__ v8f wmma_f16(v16h a, v16h b, v8f c) {
    return __builtin_amdgcn_wmma_f32_16x16x32_f16(false, a, false, b, (short)0, c, false, false);
}

// Butterfly reductions within each 16-lane half (masks 1..8 never cross the
// half boundary, matching the WMMA C-layout row split).
__device__ __forceinline__ float red_max16(float v) {
#pragma unroll
    for (int m = 1; m <= 8; m <<= 1) v = fmaxf(v, __shfl_xor(v, m, 32));
    return v;
}
__device__ __forceinline__ float red_sum16(float v) {
#pragma unroll
    for (int m = 1; m <= 8; m <<= 1) v += __shfl_xor(v, m, 32);
    return v;
}

// ---------------- fp32 -> fp16 conversion ----------------
__global__ void cvt_f32_to_f16(const float* __restrict__ in, _Float16* __restrict__ out, int n) {
    for (int i = blockIdx.x * blockDim.x + threadIdx.x; i < n; i += gridDim.x * blockDim.x)
        out[i] = (_Float16)in[i];
}

// ---------------- QKV GEMM: [8192,512] @ Wt[1536,512] + b ----------------
// 32x64 tile per wave. Epilogue scatters: Q,K -> [BH][N][D] ; V -> [BH][D][N]
__global__ __launch_bounds__(256) void gemm_qkv(
    const _Float16* __restrict__ A,   // xh [8192,512]
    const _Float16* __restrict__ Wt,  // qkv_w f16 [1536,512]
    const float*    __restrict__ bias,
    _Float16* __restrict__ qh, _Float16* __restrict__ kh, _Float16* __restrict__ vth) {
    const int wave = threadIdx.x >> 5, lane = threadIdx.x & 31;
    const int half = lane >> 4, lr = lane & 15;
    const int nGroups = QKVN / 64;                 // 24
    const int gw = blockIdx.x * 8 + wave;
    const int m0 = (gw / nGroups) * 32;
    const int n0 = (gw % nGroups) * 64;

    v8f acc[2][4] = {};
    for (int k0 = 0; k0 < CDIM; k0 += 32) {
        v16h a0 = load_a_frag(A + (size_t)m0 * CDIM + k0, CDIM, lane);
        v16h a1 = load_a_frag(A + (size_t)(m0 + 16) * CDIM + k0, CDIM, lane);
        v16h b0 = load_b_frag(Wt + (size_t)(n0 +  0) * CDIM + k0, CDIM, lane);
        v16h b1 = load_b_frag(Wt + (size_t)(n0 + 16) * CDIM + k0, CDIM, lane);
        v16h b2 = load_b_frag(Wt + (size_t)(n0 + 32) * CDIM + k0, CDIM, lane);
        v16h b3 = load_b_frag(Wt + (size_t)(n0 + 48) * CDIM + k0, CDIM, lane);
        acc[0][0] = wmma_f16(a0, b0, acc[0][0]);
        acc[0][1] = wmma_f16(a0, b1, acc[0][1]);
        acc[0][2] = wmma_f16(a0, b2, acc[0][2]);
        acc[0][3] = wmma_f16(a0, b3, acc[0][3]);
        acc[1][0] = wmma_f16(a1, b0, acc[1][0]);
        acc[1][1] = wmma_f16(a1, b1, acc[1][1]);
        acc[1][2] = wmma_f16(a1, b2, acc[1][2]);
        acc[1][3] = wmma_f16(a1, b3, acc[1][3]);
    }
#pragma unroll
    for (int mt = 0; mt < 2; ++mt) {
#pragma unroll
        for (int nt = 0; nt < 4; ++nt) {
            const int c   = n0 + nt * 16 + lr;
            const int sec = c >> 9;            // 0=q 1=k 2=v
            const int cc  = c & (CDIM - 1);
            const int h   = cc >> 7;
            const int d   = cc & (HDIM - 1);
            const float bv = bias[c];
#pragma unroll
            for (int r = 0; r < 8; ++r) {
                const int m = m0 + mt * 16 + r + half * 8;
                const int b = m >> 12;         // /4096
                const int n = m & (NTOK - 1);
                const _Float16 hv = (_Float16)(acc[mt][nt][r] + bv);
                const size_t bh = (size_t)(b * NHEAD + h);
                if (sec == 0)      qh[(bh * NTOK + n) * HDIM + d] = hv;
                else if (sec == 1) kh[(bh * NTOK + n) * HDIM + d] = hv;
                else               vth[(bh * HDIM + d) * NTOK + n] = hv;
            }
        }
    }
}

// ---------------- Flash attention ----------------
// grid(B*H, N/64), 128 threads (4 waves), wave owns 16 query rows.
// Softmax state (m, l, alpha) lives in registers, reduced with shfl butterflies;
// LDS only holds the 16x64 f16 P tile for the C->A layout transpose.
__global__ __launch_bounds__(128) void flash_attn(
    const _Float16* __restrict__ qh,   // [BH][N][D]
    const _Float16* __restrict__ kh,   // [BH][N][D]
    const _Float16* __restrict__ vth,  // [BH][D][N]
    _Float16* __restrict__ oh) {       // [B][N][C]
    __shared__ _Float16 Psc[4][16][64];

    const int wave = threadIdx.x >> 5, lane = threadIdx.x & 31;
    const int half = lane >> 4, lr = lane & 15;
    const int bh = blockIdx.x;
    const int b = bh >> 2, h = bh & (NHEAD - 1);
    const int m0 = blockIdx.y * 64 + wave * 16;

    const _Float16* qb  = qh  + (size_t)bh * NTOK * HDIM;
    const _Float16* kb  = kh  + (size_t)bh * NTOK * HDIM;
    const _Float16* vtb = vth + (size_t)bh * HDIM * NTOK;
    _Float16* pr = &Psc[wave][0][0];

    v16h qf[4];
#pragma unroll
    for (int kk = 0; kk < 4; ++kk)
        qf[kk] = load_a_frag(qb + (size_t)m0 * HDIM + kk * 32, HDIM, lane);

    v8f oacc[8] = {};
    float mrow[8], lrow[8];
#pragma unroll
    for (int r = 0; r < 8; ++r) { mrow[r] = -1e30f; lrow[r] = 0.f; }

    // (1/sqrt(128)) * log2(e): softmax runs in exp2 domain (raw v_exp_f32)
    const float scale2 = 0.12751211752736637f;

    for (int j = 0; j < NTOK; j += 64) {
        // prefetch next KV block while this one computes
        if (j + 64 < NTOK) {
            __builtin_prefetch(kb + (size_t)(j + 64 + lane * 2) * HDIM, 0, 3);
            __builtin_prefetch(vtb + (size_t)(lane * 4) * NTOK + (j + 64), 0, 3);
        }

        // ---- S = Q K^T (16x64), all 4 B-frags loaded before the WMMA group ----
        v8f sacc[4] = {};
#pragma unroll
        for (int kk = 0; kk < 4; ++kk) {
            v16h k0f = load_b_frag(kb + (size_t)(j +  0) * HDIM + kk * 32, HDIM, lane);
            v16h k1f = load_b_frag(kb + (size_t)(j + 16) * HDIM + kk * 32, HDIM, lane);
            v16h k2f = load_b_frag(kb + (size_t)(j + 32) * HDIM + kk * 32, HDIM, lane);
            v16h k3f = load_b_frag(kb + (size_t)(j + 48) * HDIM + kk * 32, HDIM, lane);
            sacc[0] = wmma_f16(qf[kk], k0f, sacc[0]);
            sacc[1] = wmma_f16(qf[kk], k1f, sacc[1]);
            sacc[2] = wmma_f16(qf[kk], k2f, sacc[2]);
            sacc[3] = wmma_f16(qf[kk], k3f, sacc[3]);
        }

        // ---- register-resident online softmax ----
        float al[8];
#pragma unroll
        for (int r = 0; r < 8; ++r) {
#pragma unroll
            for (int nt = 0; nt < 4; ++nt) sacc[nt][r] *= scale2;
            float pm = fmaxf(fmaxf(sacc[0][r], sacc[1][r]), fmaxf(sacc[2][r], sacc[3][r]));
            const float bm = red_max16(pm);
            const float nm = fmaxf(mrow[r], bm);
            al[r] = EXP2F(mrow[r] - nm);
            mrow[r] = nm;
            float ps = 0.f;
#pragma unroll
            for (int nt = 0; nt < 4; ++nt) {
                const float p = EXP2F(sacc[nt][r] - nm);
                ps += p;
                Psc[wave][r + half * 8][nt * 16 + lr] = (_Float16)p;
            }
            lrow[r] = lrow[r] * al[r] + red_sum16(ps);
        }
        // per-wave-private LDS: same-wave DS ops are in-order, no barrier needed

        // ---- O = diag(alpha)*O + P V ----
        v16h pf0 = load_a_frag(pr, 64, lane);
        v16h pf1 = load_a_frag(pr + 32, 64, lane);
#pragma unroll
        for (int dt = 0; dt < 8; ++dt) {
            v16h vf0 = load_b_frag(vtb + (size_t)(dt * 16) * NTOK + j, NTOK, lane);
            v16h vf1 = load_b_frag(vtb + (size_t)(dt * 16) * NTOK + j + 32, NTOK, lane);
#pragma unroll
            for (int r = 0; r < 8; ++r) oacc[dt][r] *= al[r];
            oacc[dt] = wmma_f16(pf0, vf0, oacc[dt]);
            oacc[dt] = wmma_f16(pf1, vf1, oacc[dt]);
        }
    }

    // ---- epilogue: O /= l, store f16 to [B][N][C] ----
    float linv[8];
#pragma unroll
    for (int r = 0; r < 8; ++r) linv[r] = 1.0f / lrow[r];
#pragma unroll
    for (int dt = 0; dt < 8; ++dt)
#pragma unroll
        for (int r = 0; r < 8; ++r) {
            const int nq = m0 + r + half * 8;
            const int c  = h * HDIM + dt * 16 + lr;
            oh[((size_t)(b * NTOK + nq)) * CDIM + c] = (_Float16)(oacc[dt][r] * linv[r]);
        }
}

// ---------------- output projection: [8192,512] @ Wt[512,512] + b -> f32 ----
__global__ __launch_bounds__(256) void gemm_proj(
    const _Float16* __restrict__ A,   // oh [8192,512]
    const _Float16* __restrict__ Wt,  // proj_w f16 [512,512]
    const float*    __restrict__ bias,
    float* __restrict__ out) {
    const int wave = threadIdx.x >> 5, lane = threadIdx.x & 31;
    const int half = lane >> 4, lr = lane & 15;
    const int nGroups = CDIM / 64;                 // 8
    const int gw = blockIdx.x * 8 + wave;
    const int m0 = (gw / nGroups) * 32;
    const int n0 = (gw % nGroups) * 64;

    v8f acc[2][4] = {};
    for (int k0 = 0; k0 < CDIM; k0 += 32) {
        v16h a0 = load_a_frag(A + (size_t)m0 * CDIM + k0, CDIM, lane);
        v16h a1 = load_a_frag(A + (size_t)(m0 + 16) * CDIM + k0, CDIM, lane);
        v16h b0 = load_b_frag(Wt + (size_t)(n0 +  0) * CDIM + k0, CDIM, lane);
        v16h b1 = load_b_frag(Wt + (size_t)(n0 + 16) * CDIM + k0, CDIM, lane);
        v16h b2 = load_b_frag(Wt + (size_t)(n0 + 32) * CDIM + k0, CDIM, lane);
        v16h b3 = load_b_frag(Wt + (size_t)(n0 + 48) * CDIM + k0, CDIM, lane);
        acc[0][0] = wmma_f16(a0, b0, acc[0][0]);
        acc[0][1] = wmma_f16(a0, b1, acc[0][1]);
        acc[0][2] = wmma_f16(a0, b2, acc[0][2]);
        acc[0][3] = wmma_f16(a0, b3, acc[0][3]);
        acc[1][0] = wmma_f16(a1, b0, acc[1][0]);
        acc[1][1] = wmma_f16(a1, b1, acc[1][1]);
        acc[1][2] = wmma_f16(a1, b2, acc[1][2]);
        acc[1][3] = wmma_f16(a1, b3, acc[1][3]);
    }
#pragma unroll
    for (int mt = 0; mt < 2; ++mt)
#pragma unroll
        for (int nt = 0; nt < 4; ++nt) {
            const int c = n0 + nt * 16 + lr;
            const float bv = bias[c];
#pragma unroll
            for (int r = 0; r < 8; ++r) {
                const int m = m0 + mt * 16 + r + half * 8;
                out[(size_t)m * CDIM + c] = acc[mt][nt][r] + bv;
            }
        }
}

// ---------------- host launcher ----------------
extern "C" void kernel_launch(void* const* d_in, const int* in_sizes, int n_in,
                              void* d_out, int out_size, void* d_ws, size_t ws_size,
                              hipStream_t stream) {
    const float* x      = (const float*)d_in[0];
    const float* qkv_w  = (const float*)d_in[1];
    const float* qkv_b  = (const float*)d_in[2];
    const float* proj_w = (const float*)d_in[3];
    const float* proj_b = (const float*)d_in[4];
    float* out = (float*)d_out;

    // workspace layout (all f16)
    char* ws = (char*)d_ws;
    size_t off = 0;
    auto alloc = [&](size_t bytes) { char* p = ws + off; off += (bytes + 255) & ~(size_t)255; return p; };
    _Float16* xh  = (_Float16*)alloc((size_t)MROWS * CDIM * 2);
    _Float16* qwh = (_Float16*)alloc((size_t)QKVN * CDIM * 2);
    _Float16* pwh = (_Float16*)alloc((size_t)CDIM * CDIM * 2);
    _Float16* qh  = (_Float16*)alloc((size_t)BATCH * NHEAD * NTOK * HDIM * 2);
    _Float16* kh  = (_Float16*)alloc((size_t)BATCH * NHEAD * NTOK * HDIM * 2);
    _Float16* vth = (_Float16*)alloc((size_t)BATCH * NHEAD * HDIM * NTOK * 2);
    _Float16* oh  = (_Float16*)alloc((size_t)MROWS * CDIM * 2);
    (void)ws_size;

    cvt_f32_to_f16<<<4096, 256, 0, stream>>>(x, xh, MROWS * CDIM);
    cvt_f32_to_f16<<<1024, 256, 0, stream>>>(qkv_w, qwh, QKVN * CDIM);
    cvt_f32_to_f16<<<512, 256, 0, stream>>>(proj_w, pwh, CDIM * CDIM);

    // (8192/32) * (1536/64) waves / 8 per block = 768 blocks
    gemm_qkv<<<768, 256, 0, stream>>>(xh, qwh, qkv_b, qh, kh, vth);

    dim3 agrid(BATCH * NHEAD, NTOK / 64);
    flash_attn<<<agrid, 128, 0, stream>>>(qh, kh, vth, oh);

    // (8192/32) * (512/64) / 8 = 256 blocks
    gemm_proj<<<256, 256, 0, stream>>>(oh, pwh, proj_b, out);
}
